// FeedForwardQuantum_65481071410316
// MI455X (gfx1250) — compile-verified, matrix operand order checked by reading, not run on hardware
//
#include <hip/hip_runtime.h>
#include <math.h>
#include <stdint.h>

typedef __attribute__((ext_vector_type(16))) _Float16 v16h;
typedef __attribute__((ext_vector_type(8)))  _Float16 v8h;
typedef __attribute__((ext_vector_type(8)))  float    v8f;
typedef __attribute__((ext_vector_type(4)))  float    v4f;
typedef __attribute__((ext_vector_type(4)))  int      i32x4;

// pointer-to-int4 in global (AS1) / LDS (AS3) address spaces, as the
// async-copy builtin expects
typedef __attribute__((address_space(1))) i32x4* global_v4i;
typedef __attribute__((address_space(3))) i32x4* lds_v4i;

#define NTOK   8192      // B*S = 4*2048
#define DMODEL 1024
#define FFDIM  4096
#define NQ     8
#define BM     128       // tokens per workgroup tile
#define BN     128       // output features per workgroup tile
#define KC     64        // K (FF) chunk per iteration
#define LDH    72        // padded f16 stride (144B, 16B aligned, conflict-free)
#define NPREF  4         // async b128 issues per thread per chunk

#if defined(__gfx1250__) && __has_builtin(__builtin_amdgcn_global_load_async_to_lds_b128)
#define HAS_ASYNC_LDS 1
#else
#define HAS_ASYNC_LDS 0
#endif

#if HAS_ASYNC_LDS
#if __has_builtin(__builtin_amdgcn_s_wait_asynccnt)
#define WAIT_ASYNC(n) __builtin_amdgcn_s_wait_asynccnt(n)
#else
#define WAIT_ASYNC(n) asm volatile("s_wait_asynccnt %0" ::"n"(n) : "memory")
#endif

__device__ __forceinline__ void issue_w2_async(const _Float16* __restrict__ w2h,
                                               _Float16 (*swb)[LDH],
                                               int n0, int kk, int tid)
{
    // 128 rows x 64 f16 = 1024 16B-chunks; 256 threads x 4 async b128 each
    #pragma unroll
    for (int i = 0; i < NPREF; ++i) {
        int idx = tid + i * 256;          // 0..1023
        int n   = idx >> 3;               // row 0..127
        int k8  = (idx & 7) * 8;          // k sub-chunk start
        const _Float16* g = w2h + (size_t)(n0 + n) * FFDIM + kk + k8;
        _Float16*       l = &swb[n][k8];
        __builtin_amdgcn_global_load_async_to_lds_b128(
            (global_v4i)(uintptr_t)g,     // global src (AS1 int4*)
            (lds_v4i)l,                   // LDS dst (AS3 int4*)
            /*imm offset*/ 0, /*cpol*/ 0);
    }
}
#endif

// ---- pre-kernel: W2 f32 -> f16 workspace (run once per launch, 8 MB) ----
__global__ __launch_bounds__(256)
void w2_to_f16_kernel(const float* __restrict__ W2, void* __restrict__ w2h_raw)
{
    _Float16* w2h = (_Float16*)w2h_raw;
    size_t i = ((size_t)blockIdx.x * 256 + threadIdx.x) * 8;
    v4f a = *(const v4f*)(W2 + i);
    v4f b = *(const v4f*)(W2 + i + 4);
    v8h o;
    o[0] = (_Float16)a.x; o[1] = (_Float16)a.y;
    o[2] = (_Float16)a.z; o[3] = (_Float16)a.w;
    o[4] = (_Float16)b.x; o[5] = (_Float16)b.y;
    o[6] = (_Float16)b.z; o[7] = (_Float16)b.w;
    *(v8h*)(w2h + i) = o;
}

template <bool USE_WS>
__global__ __launch_bounds__(256)
void ffq_fused_kernel(const float* __restrict__ x,
                      const float* __restrict__ theta,
                      const float* __restrict__ W1,
                      const float* __restrict__ b1,
                      const float* __restrict__ W2,
                      const void*  __restrict__ w2h_raw,
                      const float* __restrict__ b2,
                      float* __restrict__ out)
{
    __shared__ float    sz[BM][NQ];        // z tile (4 KB)
    __shared__ _Float16 sh[BM][LDH];       // h chunk, f16 (18 KB)
    __shared__ _Float16 sw[2][BN][LDH];    // W2 chunk, f16, double-buffered (36 KB)

    const _Float16* w2h = (const _Float16*)w2h_raw;

    const int tid   = threadIdx.x;
    const int m0    = blockIdx.x * BM;
    const int n0    = blockIdx.y * BN;
    const int lane  = tid & 31;
    const int wave  = tid >> 5;
    const int wm    = (wave & 1) * 64;
    const int wn    = (wave >> 1) * 32;
    const int lrow  = lane & 15;
    const int khalf = lane >> 4;

    // kick off async copy of chunk 0 before anything else
    if constexpr (USE_WS) {
#if HAS_ASYNC_LDS
        issue_w2_async(w2h, sw[0], n0, 0, tid);
#endif
    }

    // ---- 1) z tile: z[m][q] = cos(x[m0+m, q]) * cos(theta[q]) ----
    #pragma unroll
    for (int i = 0; i < 4; ++i) {
        int idx = tid + i * 256;                 // 0..1023
        int m = idx >> 3, q = idx & 7;
        float xv = x[(size_t)(m0 + m) * DMODEL + q];
        sz[m][q] = cosf(xv) * cosf(theta[q]);
    }

    v8f acc[4][2];
    #pragma unroll
    for (int mt = 0; mt < 4; ++mt)
        #pragma unroll
        for (int nt = 0; nt < 2; ++nt)
            #pragma unroll
            for (int e = 0; e < 8; ++e)
                acc[mt][nt][e] = 0.0f;

    const int kl     = tid & 63;          // this thread's k within chunk
    const int mstart = (tid >> 6) * 32;   // this thread's 32-token strip

    __syncthreads();

    int it = 0;
    for (int kk = 0; kk < FFDIM; kk += KC, ++it) {
        const int b = it & 1;

        // ---- 2a) h chunk: h[m][kl] = relu(b1 + z[m,:8].W1[k,:8]) -> f16 LDS
        {
            const int kg = kk + kl;
            v4f w1a = *(const v4f*)(W1 + (size_t)kg * NQ);
            v4f w1b = *(const v4f*)(W1 + (size_t)kg * NQ + 4);
            float bk = b1[kg];
            #pragma unroll 4
            for (int mi = 0; mi < 32; ++mi) {
                int m = mstart + mi;
                v4f z0 = *(const v4f*)(&sz[m][0]);   // lane-broadcast reads
                v4f z1 = *(const v4f*)(&sz[m][4]);
                float a = bk;
                a += z0.x * w1a.x + z0.y * w1a.y + z0.z * w1a.z + z0.w * w1a.w;
                a += z1.x * w1b.x + z1.y * w1b.y + z1.z * w1b.z + z1.w * w1b.w;
                sh[m][kl] = (_Float16)fmaxf(a, 0.0f);
            }
        }

        // ---- 2b) W2 chunk staging ----
        if constexpr (USE_WS) {
#if HAS_ASYNC_LDS
            if (kk + KC < FFDIM) {
                issue_w2_async(w2h, sw[b ^ 1], n0, kk + KC, tid);  // prefetch next
                WAIT_ASYNC(NPREF);   // previous chunk's copies (in-order) done
            } else {
                WAIT_ASYNC(0);
            }
#else
            #pragma unroll
            for (int i = 0; i < 4; ++i) {
                int idx = tid + i * 256;
                int n = idx >> 3, k8 = (idx & 7) * 8;
                *(v8h*)&sw[b][n][k8] =
                    *(const v8h*)(w2h + (size_t)(n0 + n) * FFDIM + kk + k8);
            }
#endif
        } else {
            // fallback: stage from f32 W2 with in-loop conversion
            #pragma unroll
            for (int i = 0; i < 32; ++i) {
                int idx = tid + i * 256;
                int n = idx >> 6, k = idx & 63;
                sw[b][n][k] = (_Float16)W2[(size_t)(n0 + n) * FFDIM + kk + k];
            }
        }
        __syncthreads();

        // ---- 3) WMMA: 2 K-steps of 32, 4x2 tiles per wave ----
        #pragma unroll
        for (int ks = 0; ks < 2; ++ks) {
            v16h afrag[4], bfrag[2];
            #pragma unroll
            for (int mt = 0; mt < 4; ++mt) {
                const _Float16* p = &sh[wm + mt * 16 + lrow][ks * 32 + khalf * 8];
                v8h lo = *(const v8h*)p;
                v8h hi = *(const v8h*)(p + 16);
                afrag[mt] = __builtin_shufflevector(lo, hi,
                    0, 1, 2, 3, 4, 5, 6, 7, 8, 9, 10, 11, 12, 13, 14, 15);
            }
            #pragma unroll
            for (int nt = 0; nt < 2; ++nt) {
                const _Float16* p = &sw[b][wn + nt * 16 + lrow][ks * 32 + khalf * 8];
                v8h lo = *(const v8h*)p;
                v8h hi = *(const v8h*)(p + 16);
                bfrag[nt] = __builtin_shufflevector(lo, hi,
                    0, 1, 2, 3, 4, 5, 6, 7, 8, 9, 10, 11, 12, 13, 14, 15);
            }
            #pragma unroll
            for (int mt = 0; mt < 4; ++mt)
                #pragma unroll
                for (int nt = 0; nt < 2; ++nt)
                    acc[mt][nt] = __builtin_amdgcn_wmma_f32_16x16x32_f16(
                        false, afrag[mt], false, bfrag[nt],
                        (short)0, acc[mt][nt], false, false);
        }
        __syncthreads();
    }

    // ---- 4) epilogue: out = acc + b2 (C layout: lane=N, VGPR r -> M) ----
    #pragma unroll
    for (int nt = 0; nt < 2; ++nt) {
        int n = n0 + wn + nt * 16 + lrow;
        float bias = b2[n];
        #pragma unroll
        for (int mt = 0; mt < 4; ++mt) {
            int mbase = m0 + wm + mt * 16 + khalf * 8;
            #pragma unroll
            for (int r = 0; r < 8; ++r) {
                out[(size_t)(mbase + r) * DMODEL + n] = acc[mt][nt][r] + bias;
            }
        }
    }
}

extern "C" void kernel_launch(void* const* d_in, const int* in_sizes, int n_in,
                              void* d_out, int out_size, void* d_ws, size_t ws_size,
                              hipStream_t stream) {
    const float* x     = (const float*)d_in[0];
    const float* theta = (const float*)d_in[1];
    const float* W1    = (const float*)d_in[2];
    const float* b1    = (const float*)d_in[3];
    const float* W2    = (const float*)d_in[4];
    const float* b2    = (const float*)d_in[5];
    float* out = (float*)d_out;

    dim3 grid(NTOK / BM, DMODEL / BN);   // 64 x 8 = 512 workgroups
    const size_t w2h_bytes = (size_t)DMODEL * FFDIM * 2;   // 8 MB f16 W2

    if (ws_size >= w2h_bytes) {
        // convert W2 to f16 once, then run the async double-buffered kernel
        int nblk = (DMODEL * FFDIM) / (256 * 8);           // 2048
        w2_to_f16_kernel<<<nblk, 256, 0, stream>>>(W2, d_ws);
        ffq_fused_kernel<true><<<grid, 256, 0, stream>>>(
            x, theta, W1, b1, W2, d_ws, b2, out);
    } else {
        ffq_fused_kernel<false><<<grid, 256, 0, stream>>>(
            x, theta, W1, b1, W2, nullptr, b2, out);
    }
}
